// RegionalEncoder_90305982366363
// MI455X (gfx1250) — matrix operand order repaired and checked by reading, not compiled, not run on hardware
//
#include <hip/hip_runtime.h>
#include <hip/hip_bf16.h>

// ---------------------------------------------------------------------------
// RegionalEncoder for MI455X (gfx1250, wave32):
//   x = silu(X @ W1 + b1) @ W2 + b2      (bf16 WMMA, fp32 accumulate)
//   out = scatter_mean(x[src], dst)      (f32 global atomics)
// ---------------------------------------------------------------------------

typedef __attribute__((ext_vector_type(16))) __bf16         v16bf;
typedef __attribute__((ext_vector_type(8)))  float          v8f;
typedef __attribute__((ext_vector_type(8)))  unsigned short u16x8;
typedef __attribute__((ext_vector_type(16))) unsigned short u16x16;

#define D        512                // input_dim == hidden_dim
#define KT_STEPS (D / 32)           // 16 WMMA K-steps per GEMM
#define BROWS    32                 // rows of X per workgroup (2 M-tiles)
#define LDH      (D + 8)            // padded LDS row stride (bf16 elems) -> conflict-free

__device__ __forceinline__ unsigned short f2bf(float f) {
  unsigned int u = __float_as_uint(f);
  u += 0x7FFFu + ((u >> 16) & 1u);      // round-to-nearest-even
  return (unsigned short)(u >> 16);
}

// silu(x) = x * sigmoid(x); use v_rcp_f32 instead of IEEE div (avoids the
// v_div_scale/v_div_fmas fixup sequence seen in the previous disassembly).
__device__ __forceinline__ float silu_fast(float v) {
  float e = __expf(-v);                          // v_exp_f32
  return v * __builtin_amdgcn_rcpf(1.0f + e);    // v_rcp_f32
}

__device__ __forceinline__ v8f wmma_bf16(u16x16 a, u16x16 b, v8f c) {
  return __builtin_amdgcn_wmma_f32_16x16x32_bf16(
      false, __builtin_bit_cast(v16bf, a),
      false, __builtin_bit_cast(v16bf, b),
      (short)0, c, false, false);
}

// A fragment (16x32 bf16) from LDS. ISA layout: lane L -> row M = L&15;
// VGPR0..3 hold K = kb..kb+7, VGPR4..7 hold K = kb+16..kb+23, kb = (L>>4)*8.
__device__ __forceinline__ u16x16 load_a(const unsigned short* sh, int mt, int kt,
                                         int mlane, int hsel) {
  const unsigned short* p = sh + (mt * 16 + mlane) * LDH + kt * 32 + hsel * 8;
  u16x8 lo = *(const u16x8*)p;          // ds_load_b128
  u16x8 hi = *(const u16x8*)(p + 16);   // ds_load_b128
  u16x16 r;
#pragma unroll
  for (int t = 0; t < 8; ++t) { r[t] = lo[t]; r[t + 8] = hi[t]; }
  return r;
}

// B fragment (32x16 bf16) from pre-swizzled weights: element e of lane L is
// W[kt*32 + (L>>4)*8 + (e&7) + (e>>3)*16][nt*16 + (L&15)], stored contiguously per lane.
__device__ __forceinline__ u16x16 load_b(const unsigned short* __restrict__ wp,
                                         int nt, int kt, int lane) {
  const unsigned short* p = wp + (((size_t)(nt * KT_STEPS + kt) * 32 + lane) << 4);
  u16x8 lo = *(const u16x8*)p;          // global_load_b128 (coalesced, L2-hit)
  u16x8 hi = *(const u16x8*)(p + 8);
  u16x16 r;
#pragma unroll
  for (int t = 0; t < 8; ++t) { r[t] = lo[t]; r[t + 8] = hi[t]; }
  return r;
}

// --------------------- weight pack: fp32 -> bf16 fragment order -------------
__global__ __launch_bounds__(256) void pack_w_kernel(const float* __restrict__ w,
                                                     unsigned short* __restrict__ wp) {
  int i = blockIdx.x * 256 + threadIdx.x;          // over 512*512
  if (i >= D * D) return;
  int e    = i & 15;
  int lane = (i >> 4) & 31;
  int kt   = (i >> 9) & (KT_STEPS - 1);
  int nt   = i >> 13;
  int k = kt * 32 + (lane >> 4) * 8 + (e & 7) + (e >> 3) * 16;
  int n = nt * 16 + (lane & 15);
  wp[i] = f2bf(w[k * D + n]);
}

// --------------------- fused MLP: GEMM -> SiLU -> GEMM ----------------------
__global__ __launch_bounds__(256) void mlp_kernel(
    const float* __restrict__ X, const unsigned short* __restrict__ w1p,
    const float* __restrict__ b1, const unsigned short* __restrict__ w2p,
    const float* __restrict__ b2, float* __restrict__ xout, int nrows) {
  __shared__ __align__(16) unsigned short sh[BROWS * LDH];   // 33.3 KB

  const int lane  = threadIdx.x & 31;
  const int wave  = threadIdx.x >> 5;      // 8 waves; wave owns 4 N-tiles
  const int row0  = blockIdx.x * BROWS;
  const int mlane = lane & 15;
  const int hsel  = lane >> 4;

  // Warm L2/L0 for the (small, reused) packed weights: global_prefetch_b8.
  __builtin_prefetch(w1p + (size_t)threadIdx.x * 64, 0, 1);
  __builtin_prefetch(w2p + (size_t)threadIdx.x * 64, 0, 1);

  // Stage X block into LDS as bf16 (coalesced float4 reads).
  for (int i = threadIdx.x; i < BROWS * D / 4; i += 256) {
    int row = (i * 4) >> 9;
    int col = (i * 4) & (D - 1);
    float4 v = {0.f, 0.f, 0.f, 0.f};
    int gr = row0 + row;
    if (gr < nrows) v = *(const float4*)(X + (size_t)gr * D + col);
    unsigned short* p = &sh[row * LDH + col];
    p[0] = f2bf(v.x); p[1] = f2bf(v.y); p[2] = f2bf(v.z); p[3] = f2bf(v.w);
  }
  __syncthreads();

  v8f acc[2][4];
#pragma unroll
  for (int mt = 0; mt < 2; ++mt)
#pragma unroll
    for (int j = 0; j < 4; ++j) acc[mt][j] = (v8f)0.0f;

  // ---- GEMM1: h = X @ W1 ----
#pragma unroll 1
  for (int kt = 0; kt < KT_STEPS; ++kt) {
    u16x16 a0 = load_a(sh, 0, kt, mlane, hsel);
    u16x16 a1 = load_a(sh, 1, kt, mlane, hsel);
#pragma unroll
    for (int j = 0; j < 4; ++j) {
      u16x16 b = load_b(w1p, wave * 4 + j, kt, lane);
      acc[0][j] = wmma_bf16(a0, b, acc[0][j]);
      acc[1][j] = wmma_bf16(a1, b, acc[1][j]);
    }
  }

  // ---- bias + SiLU in registers ----
  float bias[4];
#pragma unroll
  for (int j = 0; j < 4; ++j) bias[j] = b1[(wave * 4 + j) * 16 + mlane];
#pragma unroll
  for (int mt = 0; mt < 2; ++mt)
#pragma unroll
    for (int j = 0; j < 4; ++j)
#pragma unroll
      for (int r = 0; r < 8; ++r) {
        acc[mt][j][r] = silu_fast(acc[mt][j][r] + bias[j]);
      }

  __syncthreads();   // all waves done reading X from LDS

  // Write h (D-layout -> row/col) back into the same LDS buffer as bf16.
#pragma unroll
  for (int mt = 0; mt < 2; ++mt)
#pragma unroll
    for (int j = 0; j < 4; ++j) {
      int n = (wave * 4 + j) * 16 + mlane;
#pragma unroll
      for (int r = 0; r < 8; ++r) {
        int m = mt * 16 + hsel * 8 + r;
        sh[m * LDH + n] = f2bf(acc[mt][j][r]);
      }
    }
  __syncthreads();

  // ---- GEMM2: x = h @ W2 ----
#pragma unroll
  for (int mt = 0; mt < 2; ++mt)
#pragma unroll
    for (int j = 0; j < 4; ++j) acc[mt][j] = (v8f)0.0f;

#pragma unroll 1
  for (int kt = 0; kt < KT_STEPS; ++kt) {
    u16x16 a0 = load_a(sh, 0, kt, mlane, hsel);
    u16x16 a1 = load_a(sh, 1, kt, mlane, hsel);
#pragma unroll
    for (int j = 0; j < 4; ++j) {
      u16x16 b = load_b(w2p, wave * 4 + j, kt, lane);
      acc[0][j] = wmma_bf16(a0, b, acc[0][j]);
      acc[1][j] = wmma_bf16(a1, b, acc[1][j]);
    }
  }

  // ---- bias2 + store fp32 rows ----
#pragma unroll
  for (int j = 0; j < 4; ++j) bias[j] = b2[(wave * 4 + j) * 16 + mlane];
#pragma unroll
  for (int mt = 0; mt < 2; ++mt)
#pragma unroll
    for (int j = 0; j < 4; ++j) {
      int n = (wave * 4 + j) * 16 + mlane;
#pragma unroll
      for (int r = 0; r < 8; ++r) {
        int gr = row0 + mt * 16 + hsel * 8 + r;
        if (gr < nrows) xout[(size_t)gr * D + n] = acc[mt][j][r] + bias[j];
      }
    }
}

// --------------------- scatter-mean phase -----------------------------------
__global__ __launch_bounds__(256) void zero_kernel(float* __restrict__ out, int nout,
                                                   float* __restrict__ cnt, int nmesh) {
  int i = blockIdx.x * 256 + threadIdx.x;
  if (i < nout) out[i] = 0.f;
  if (i < nmesh) cnt[i] = 0.f;
}

__global__ __launch_bounds__(256) void count_kernel(const int* __restrict__ edges,
                                                    int E, float* __restrict__ cnt) {
  int e = blockIdx.x * 256 + threadIdx.x;
  if (e < E) atomicAdd(&cnt[edges[E + e]], 1.0f);
}

__global__ __launch_bounds__(256) void scatter_kernel(const float* __restrict__ x,
                                                      const int* __restrict__ edges,
                                                      int E, float* __restrict__ out) {
  int e = blockIdx.x;               // one edge per block
  int src = edges[e];
  int dst = edges[E + e];
  const float* xr = x + (size_t)src * D;
  float* orow = out + (size_t)dst * D;
  int c = threadIdx.x;
  atomicAdd(&orow[c],       xr[c]);         // global_atomic_add_f32
  atomicAdd(&orow[c + 256], xr[c + 256]);
}

__global__ __launch_bounds__(256) void finalize_kernel(float* __restrict__ out,
                                                       const float* __restrict__ cnt,
                                                       int nout) {
  int i = blockIdx.x * 256 + threadIdx.x;
  if (i < nout) out[i] = out[i] / fmaxf(cnt[i >> 9], 1.0f);
}

// ---------------------------------------------------------------------------
extern "C" void kernel_launch(void* const* d_in, const int* in_sizes, int n_in,
                              void* d_out, int out_size, void* d_ws, size_t ws_size,
                              hipStream_t stream) {
  const float* X  = (const float*)d_in[0];
  const float* w1 = (const float*)d_in[1];
  const float* b1 = (const float*)d_in[2];
  const float* w2 = (const float*)d_in[3];
  const float* b2 = (const float*)d_in[4];
  const int* edges = (const int*)d_in[5];

  const int nrows = in_sizes[0] / D;      // 100000
  const int E     = in_sizes[5] / 2;      // 150000
  const int nmesh = out_size / D;         // 10000
  float* out = (float*)d_out;

  // Workspace layout: [w1 packed bf16][w2 packed bf16][x fp32][counts]
  unsigned char* ws = (unsigned char*)d_ws;
  unsigned short* w1p = (unsigned short*)ws;
  unsigned short* w2p = (unsigned short*)(ws + (size_t)D * D * 2);
  float* xbuf = (float*)(ws + 2 * (size_t)D * D * 2);
  float* cnt  = (float*)(ws + 2 * (size_t)D * D * 2 + (size_t)nrows * D * sizeof(float));

  const int packBlocks = (D * D + 255) / 256;
  pack_w_kernel<<<packBlocks, 256, 0, stream>>>(w1, w1p);
  pack_w_kernel<<<packBlocks, 256, 0, stream>>>(w2, w2p);

  mlp_kernel<<<(nrows + BROWS - 1) / BROWS, 256, 0, stream>>>(
      X, w1p, b1, w2p, b2, xbuf, nrows);

  zero_kernel<<<(out_size + 255) / 256, 256, 0, stream>>>(out, out_size, cnt, nmesh);
  count_kernel<<<(E + 255) / 256, 256, 0, stream>>>(edges, E, cnt);
  scatter_kernel<<<E, 256, 0, stream>>>(xbuf, edges, E, out);
  finalize_kernel<<<(out_size + 255) / 256, 256, 0, stream>>>(out, cnt, out_size);
}
